// SlidingWindowAttention_23957327577817
// MI455X (gfx1250) — matrix operand order repaired and checked by reading, not compiled, not run on hardware
//
#include <hip/hip_runtime.h>
#include <hip/hip_bf16.h>

typedef _Float16 h16;
typedef __attribute__((ext_vector_type(16))) _Float16 v16h;
typedef __attribute__((ext_vector_type(8)))  _Float16 v8h;
typedef __attribute__((ext_vector_type(8)))  float    v8f;

#define NB   2
#define NS   2048
#define NE   1024
#define NH   16
#define NKV  4
#define ND   64
#define NW   512
#define NREPC (NH / NKV)

#define MASKV (-3.0e38f)

// ---------------------------------------------------------------------------
// f32 -> f16 conversion
// ---------------------------------------------------------------------------
__global__ __launch_bounds__(256) void cvt_f32_f16(const float* __restrict__ s,
                                                   h16* __restrict__ d, int n) {
  int i = blockIdx.x * blockDim.x + threadIdx.x;
  if (i < n) d[i] = (h16)s[i];
}

// ---------------------------------------------------------------------------
// WMMA GEMM: C[M,N] = A[M,K] * Bw[N,K]^T  (f16 in, f32 out)
// one wave per 16(M) x 64(N) strip; K multiple of 32
// A-layout per lane: row = m0+lid, 16 contiguous halves at k0+hid*16
// B-layout per lane: col = n0+t*16+lid, 16 contiguous halves at k0+hid*16
// ---------------------------------------------------------------------------
__global__ __launch_bounds__(32) void wmma_gemm_nt(const h16* __restrict__ A,
                                                   const h16* __restrict__ Bw,
                                                   float* __restrict__ C,
                                                   int M, int N, int K) {
  const int lane = threadIdx.x & 31;
  const int lid  = lane & 15;
  const int hid  = lane >> 4;
  const int n0 = blockIdx.x * 64;
  const int m0 = blockIdx.y * 16;

  v8f acc[4] = {};
  const h16* arow = A + (size_t)(m0 + lid) * K + hid * 16;

  for (int k0 = 0; k0 < K; k0 += 32) {
    __builtin_prefetch(arow + k0 + 128, 0, 1);
    v8h lo = *(const v8h*)(arow + k0);
    v8h hi = *(const v8h*)(arow + k0 + 8);
    v16h a;
#pragma unroll
    for (int e = 0; e < 8; ++e) { a[e] = lo[e]; a[e + 8] = hi[e]; }
#pragma unroll
    for (int t = 0; t < 4; ++t) {
      const h16* bp = Bw + (size_t)(n0 + t * 16 + lid) * K + k0 + hid * 16;
      v16h b = *(const v16h*)bp;
      acc[t] = __builtin_amdgcn_wmma_f32_16x16x32_f16(
          false, a, false, b, (short)0, acc[t], false, false);
    }
  }

#pragma unroll
  for (int t = 0; t < 4; ++t)
#pragma unroll
    for (int r = 0; r < 8; ++r)
      C[(size_t)(m0 + r + 8 * hid) * N + n0 + t * 16 + lid] = acc[t][r];
}

// ---------------------------------------------------------------------------
// Fused RMSNorm + RoPE over one D=64 head-row per wave; f32 in, f16 out.
// lane l owns the RoPE pair (2l, 2l+1).
// ---------------------------------------------------------------------------
__global__ __launch_bounds__(256) void norm_rope_kernel(
    const float* __restrict__ src, const float* __restrict__ w,
    h16* __restrict__ dst, int nheads, int nrows) {
  int wv   = blockIdx.x * (blockDim.x >> 5) + (threadIdx.x >> 5);
  int lane = threadIdx.x & 31;
  if (wv >= nrows) return;
  int spos = (wv / nheads) % NS;

  const float* p = src + (size_t)wv * ND;
  float xe = p[2 * lane], xo = p[2 * lane + 1];
  float ss = xe * xe + xo * xo;
#pragma unroll
  for (int m = 1; m < 32; m <<= 1) ss += __shfl_xor(ss, m, 32);
  float rn = rsqrtf(ss * (1.0f / ND) + 1e-6f);

  float a = xe * rn * w[2 * lane];
  float b = xo * rn * w[2 * lane + 1];
  float ang = (float)spos * powf(10000.0f, -(2.0f * lane) / (float)ND);
  float cs = cosf(ang), sn = sinf(ang);

  h16* q = dst + (size_t)wv * ND;
  q[2 * lane]     = (h16)(a * cs - b * sn);
  q[2 * lane + 1] = (h16)(a * sn + b * cs);
}

// ---------------------------------------------------------------------------
// V f32 [B,S,KV,D] -> f16 transposed [B,KV,D,S]
// ---------------------------------------------------------------------------
__global__ __launch_bounds__(256) void vtrans_kernel(const float* __restrict__ vf,
                                                     h16* __restrict__ vt) {
  int idx = blockIdx.x * blockDim.x + threadIdx.x;  // B*KV*D*S
  if (idx >= NB * NKV * ND * NS) return;
  int s  = idx % NS;
  int d  = (idx / NS) % ND;
  int kv = (idx / (NS * ND)) % NKV;
  int b  = idx / (NS * ND * NKV);
  vt[idx] = (h16)vf[((size_t)(b * NS + s) * NKV + kv) * ND + d];
}

// ---------------------------------------------------------------------------
// Flash-style sliding-window attention: 1 wave per (b, h, 16-query tile)
// ---------------------------------------------------------------------------
__global__ __launch_bounds__(32) void attn_kernel(const h16* __restrict__ qh,
                                                  const h16* __restrict__ kh,
                                                  const h16* __restrict__ vt,
                                                  h16* __restrict__ oh) {
  __shared__ h16 plds[16][32];

  const int lane = threadIdx.x & 31;
  const int lid  = lane & 15;
  const int hid  = lane >> 4;
  const int it = blockIdx.x, h = blockIdx.y, b = blockIdx.z;
  const int i0 = it * 16;
  const int kv = h / NREPC;

  // Q tile in A-layout: two K-halves of D=64
  v16h aq0, aq1;
  {
    const h16* qrow = qh + ((size_t)(b * NS + i0 + lid) * NH + h) * ND + hid * 16;
    v8h l0 = *(const v8h*)(qrow);
    v8h h0 = *(const v8h*)(qrow + 8);
    v8h l1 = *(const v8h*)(qrow + 32);
    v8h h1 = *(const v8h*)(qrow + 40);
#pragma unroll
    for (int e = 0; e < 8; ++e) {
      aq0[e] = l0[e]; aq0[e + 8] = h0[e];
      aq1[e] = l1[e]; aq1[e + 8] = h1[e];
    }
  }

  v8f o[4] = {};
  float mrow[8], lrow[8];
#pragma unroll
  for (int r = 0; r < 8; ++r) { mrow[r] = MASKV; lrow[r] = 0.0f; }

  int lowj = i0 - (NW - 1); if (lowj < 0) lowj = 0;
  const int jstart = lowj & ~31;
  const h16* vbase = vt + (size_t)(b * NKV + kv) * ND * NS;
  const float scale = 0.125f;  // D^-0.5

  for (int j0 = jstart; j0 <= i0 + 15; j0 += 32) {
    v8f sc[2];
#pragma unroll
    for (int u = 0; u < 2; ++u) {
      int js = j0 + u * 16;
      int jr = js + lid; if (jr > NS - 1) jr = NS - 1;  // clamp; masked below
      const h16* krow = kh + ((size_t)(b * NS + jr) * NKV + kv) * ND + hid * 16;
      v16h bk0 = *(const v16h*)(krow);
      v16h bk1 = *(const v16h*)(krow + 32);
      v8f c = {};
      c = __builtin_amdgcn_wmma_f32_16x16x32_f16(false, aq0, false, bk0,
                                                 (short)0, c, false, false);
      c = __builtin_amdgcn_wmma_f32_16x16x32_f16(false, aq1, false, bk1,
                                                 (short)0, c, false, false);
#pragma unroll
      for (int r = 0; r < 8; ++r) {
        int i = i0 + r + 8 * hid;
        int j = js + lid;
        bool ok = (j <= i) && (i - j < NW);
        c[r] = ok ? c[r] * scale : MASKV;
      }
      sc[u] = c;
    }

    // online softmax over the 32-key chunk (row = VGPR slot within half-wave)
    float alpha[8];
#pragma unroll
    for (int r = 0; r < 8; ++r) {
      float rm = fmaxf(sc[0][r], sc[1][r]);
#pragma unroll
      for (int m = 1; m < 16; m <<= 1) rm = fmaxf(rm, __shfl_xor(rm, m, 32));
      float mn = fmaxf(mrow[r], rm);
      alpha[r] = __expf(mrow[r] - mn);
      mrow[r] = mn;
      float p0 = __expf(sc[0][r] - mn);
      float p1 = __expf(sc[1][r] - mn);
      sc[0][r] = p0; sc[1][r] = p1;
      float ls = p0 + p1;
#pragma unroll
      for (int m = 1; m < 16; m <<= 1) ls += __shfl_xor(ls, m, 32);
      lrow[r] = lrow[r] * alpha[r] + ls;
    }
#pragma unroll
    for (int r = 0; r < 8; ++r) {
      o[0][r] *= alpha[r]; o[1][r] *= alpha[r];
      o[2][r] *= alpha[r]; o[3][r] *= alpha[r];
    }

    // stage P (C-layout) -> LDS -> A-layout
#pragma unroll
    for (int r = 0; r < 8; ++r) {
      int m = r + 8 * hid;
      plds[m][lid]      = (h16)sc[0][r];
      plds[m][16 + lid] = (h16)sc[1][r];
    }
    __syncthreads();  // single-wave WG: NOP, DS ordering is in-wave
    v16h ap;
    {
      v8h l0 = *(const v8h*)(&plds[lid][hid * 16]);
      v8h h0 = *(const v8h*)(&plds[lid][hid * 16 + 8]);
#pragma unroll
      for (int e = 0; e < 8; ++e) { ap[e] = l0[e]; ap[e + 8] = h0[e]; }
    }
    __syncthreads();

    // P (16x32) @ V (32x64) via transposed V (contiguous K per lane)
#pragma unroll
    for (int t = 0; t < 4; ++t) {
      const h16* vp = vbase + (size_t)(t * 16 + lid) * NS + j0 + hid * 16;
      v16h bv = *(const v16h*)vp;
      o[t] = __builtin_amdgcn_wmma_f32_16x16x32_f16(false, ap, false, bv,
                                                    (short)0, o[t], false, false);
    }
  }

#pragma unroll
  for (int t = 0; t < 4; ++t)
#pragma unroll
    for (int r = 0; r < 8; ++r) {
      int m = r + 8 * hid;
      oh[(size_t)(b * NS + i0 + m) * NE + h * ND + t * 16 + lid] =
          (h16)(o[t][r] / lrow[r]);
    }
}

// ---------------------------------------------------------------------------
extern "C" void kernel_launch(void* const* d_in, const int* in_sizes, int n_in,
                              void* d_out, int out_size, void* d_ws, size_t ws_size,
                              hipStream_t stream) {
  (void)in_sizes; (void)n_in; (void)out_size;
  const float* x  = (const float*)d_in[0];
  const float* Wq = (const float*)d_in[1];
  const float* Wk = (const float*)d_in[2];
  const float* Wv = (const float*)d_in[3];
  const float* Wo = (const float*)d_in[4];
  const float* qw = (const float*)d_in[5];
  const float* kw = (const float*)d_in[6];
  float* out = (float*)d_out;

  char* base = (char*)d_ws;
  size_t off = 0;
  auto alloc = [&](size_t bytes) -> void* {
    void* r = base + off;
    off = (off + bytes + 255) & ~(size_t)255;
    return r;
  };

  const int M = NB * NS;                        // 4096
  h16*   xh  = (h16*)alloc((size_t)M * NE * 2);
  h16*   wqh = (h16*)alloc((size_t)NE * NE * 2);
  h16*   wkh = (h16*)alloc((size_t)NKV * ND * NE * 2);
  h16*   wvh = (h16*)alloc((size_t)NKV * ND * NE * 2);
  h16*   woh = (h16*)alloc((size_t)NE * NE * 2);
  float* qf  = (float*)alloc((size_t)M * NE * 4);
  float* kf  = (float*)alloc((size_t)M * NKV * ND * 4);
  float* vf  = (float*)alloc((size_t)M * NKV * ND * 4);
  h16*   qh  = (h16*)alloc((size_t)M * NE * 2);
  h16*   kh  = (h16*)alloc((size_t)M * NKV * ND * 2);
  h16*   vt  = (h16*)alloc((size_t)NB * NKV * ND * NS * 2 + 256);  // +tail pad
  h16*   oh  = (h16*)qf;  // overlay: q f32 is dead once qh exists

  if (ws_size < off) return;  // deterministic no-op if scratch too small

  // 1) conversions
  cvt_f32_f16<<<(M * NE + 255) / 256, 256, 0, stream>>>(x, xh, M * NE);
  cvt_f32_f16<<<(NE * NE + 255) / 256, 256, 0, stream>>>(Wq, wqh, NE * NE);
  cvt_f32_f16<<<(NKV * ND * NE + 255) / 256, 256, 0, stream>>>(Wk, wkh, NKV * ND * NE);
  cvt_f32_f16<<<(NKV * ND * NE + 255) / 256, 256, 0, stream>>>(Wv, wvh, NKV * ND * NE);
  cvt_f32_f16<<<(NE * NE + 255) / 256, 256, 0, stream>>>(Wo, woh, NE * NE);

  // 2) projections
  wmma_gemm_nt<<<dim3(NE / 64, M / 16), 32, 0, stream>>>(xh, wqh, qf, M, NE, NE);
  wmma_gemm_nt<<<dim3((NKV * ND) / 64, M / 16), 32, 0, stream>>>(xh, wkh, kf, M, NKV * ND, NE);
  wmma_gemm_nt<<<dim3((NKV * ND) / 64, M / 16), 32, 0, stream>>>(xh, wvh, vf, M, NKV * ND, NE);

  // 3) RMSNorm + RoPE (q then k), f16 out
  {
    int nrq = M * NH;
    norm_rope_kernel<<<(nrq + 7) / 8, 256, 0, stream>>>(qf, qw, qh, NH, nrq);
    int nrk = M * NKV;
    norm_rope_kernel<<<(nrk + 7) / 8, 256, 0, stream>>>(kf, kw, kh, NKV, nrk);
  }

  // 4) V transpose -> [B,KV,D,S] f16
  vtrans_kernel<<<(NB * NKV * ND * NS + 255) / 256, 256, 0, stream>>>(vf, vt);

  // 5) sliding-window attention
  attn_kernel<<<dim3(NS / 16, NH, NB), 32, 0, stream>>>(qh, kh, vt, oh);

  // 6) output projection -> f32 d_out
  wmma_gemm_nt<<<dim3(NE / 64, M / 16), 32, 0, stream>>>(oh, woh, out, M, NE, NE);
}
